// SumPtrGenOutputOLD_32023276159184
// MI455X (gfx1250) — matrix operand order repaired and checked by reading, not compiled, not run on hardware
//
#include <hip/hip_runtime.h>
#include <hip/hip_bf16.h>

typedef __attribute__((ext_vector_type(16))) _Float16 v16h;
typedef __attribute__((ext_vector_type(8)))  _Float16 v8h;
typedef __attribute__((ext_vector_type(8)))  float    v8f;

#define B_DIM  1024
#define H_DIM  1024
#define V_OUT  50000
#define V_INP  32000
#define SRC    256

#define BM 128
#define BN 128
#define BK 32
#define LDK (BK + 8)   // 40 halfs = 80B rows -> 16B aligned, conflict-friendly

// ---------------- helper kernels -------------------------------------------

__global__ void k_init_inv(int* __restrict__ inv) {
    int i = blockIdx.x * 256 + threadIdx.x;
    if (i < V_OUT) inv[i] = -1;
}

__global__ void k_scatter_inv(const int* __restrict__ inp_to_act, int* __restrict__ inv) {
    int i = blockIdx.x * 256 + threadIdx.x;
    if (i < V_INP) atomicMax(&inv[inp_to_act[i]], i);   // last-write-wins == max index
}

__global__ void k_zero(float* __restrict__ p, long long n) {
    long long i = (long long)blockIdx.x * blockDim.x + threadIdx.x;
    long long stride = (long long)gridDim.x * blockDim.x;
    for (; i < n; i += stride) p[i] = 0.0f;
}

__global__ void k_scatter_add(const float* __restrict__ attn, const int* __restrict__ ctx,
                              float* __restrict__ inpdist) {
    int i = blockIdx.x * 256 + threadIdx.x;          // over B*SRC
    if (i < B_DIM * SRC) {
        int b = i >> 8;                              // SRC == 256
        atomicAdd(&inpdist[(size_t)b * V_INP + ctx[i]], attn[i]);
    }
}

// ---------------- fused GEMM (gather-on-B) + epilogue ----------------------
// Block tile 128x128; 8 waves as 4(M) x 2(N); wave patch 32x64 = 2x4 WMMA tiles.
// Grid: blockIdx.x = M (fastest) so consecutive blocks share the same W tile
// through L2 -> W streams from HBM exactly once.

__global__ __launch_bounds__(256)
void k_gemm(const float* __restrict__ x, const float* __restrict__ W,
            const float* __restrict__ bias, const int* __restrict__ out_map,
            const int* __restrict__ actionmask, const int* __restrict__ inv,
            const float* __restrict__ inpdist,
            float* __restrict__ out_scores, float* __restrict__ gen_masked) {
    __shared__ __align__(16) _Float16 As[BM][LDK];
    __shared__ __align__(16) _Float16 Bs[BN][LDK];
    __shared__ int   nrow[BN];
    __shared__ float nbias[BN];

    const int tid   = threadIdx.x;
    const int lane  = tid & 31;
    const int wave  = tid >> 5;       // 0..7
    const int waveM = wave & 3;       // 0..3 -> 32-row slab inside BM
    const int waveN = wave >> 2;      // 0..1 -> 64-col slab inside BN
    const int m0 = blockIdx.x * BM;   // M fastest-varying
    const int n0 = blockIdx.y * BN;

    if (tid < BN) {
        int j  = n0 + tid;
        int wr = (j < V_OUT) ? out_map[j] : 0;
        if (wr < 0 || wr >= V_OUT) wr = 0;
        nrow[tid]  = wr;
        nbias[tid] = (j < V_OUT) ? bias[wr] : 0.0f;
    }
    __syncthreads();

    // A loaders: 2 threads per row, 16 contiguous floats each (128x32)
    const int ar = tid >> 1;            // 0..127
    const int ac = (tid & 1) * 16;      // 0 or 16
    // B loaders: 2 threads per gathered W row, 16 contiguous floats each (128x32)
    const int bn = tid >> 1;            // 0..127
    const int bc = (tid & 1) * 16;      // 0 or 16
    const float* xrow = x + (size_t)(m0 + ar) * H_DIM + ac;
    const float* wrow = W + (size_t)nrow[bn] * H_DIM + bc;

    v8f acc[2][4] = {};

    // per-lane fragment addressing (ISA 7.12.2 layouts)
    const int akh = (lane >> 4) << 3;                    // A: K base 0 / 8
    const int bkh = (lane >> 4) << 3;                    // B: K base 0 / 8
    const _Float16* ap[2];
    const _Float16* bp[4];
#pragma unroll
    for (int mi = 0; mi < 2; ++mi)
        ap[mi] = &As[waveM * 32 + mi * 16 + (lane & 15)][0];
#pragma unroll
    for (int ni = 0; ni < 4; ++ni)
        bp[ni] = &Bs[waveN * 64 + ni * 16 + (lane & 15)][0];

    for (int k0 = 0; k0 < H_DIM; k0 += BK) {
        // stage f32 -> f16 tiles into LDS
#pragma unroll
        for (int q = 0; q < 4; ++q) {
            float4 a = *(const float4*)(xrow + k0 + 4 * q);
            As[ar][ac + 4 * q + 0] = (_Float16)a.x;
            As[ar][ac + 4 * q + 1] = (_Float16)a.y;
            As[ar][ac + 4 * q + 2] = (_Float16)a.z;
            As[ar][ac + 4 * q + 3] = (_Float16)a.w;
        }
#pragma unroll
        for (int q = 0; q < 4; ++q) {
            float4 b0 = *(const float4*)(wrow + k0 + 4 * q);
            Bs[bn][bc + 4 * q + 0] = (_Float16)b0.x;
            Bs[bn][bc + 4 * q + 1] = (_Float16)b0.y;
            Bs[bn][bc + 4 * q + 2] = (_Float16)b0.z;
            Bs[bn][bc + 4 * q + 3] = (_Float16)b0.w;
        }
        if (k0 + BK < H_DIM) {           // gfx1250 global_prefetch of next K-tile
            __builtin_prefetch(xrow + k0 + BK, 0, 3);
            __builtin_prefetch(wrow + k0 + BK, 0, 3);
        }
        __syncthreads();

        // Build 2 A frags + 4 B frags, fire 8 independent WMMAs.
        v16h af[2], bf[4];
#pragma unroll
        for (int mi = 0; mi < 2; ++mi) {
            v8h lo = *(const v8h*)(ap[mi] + akh);
            v8h hi = *(const v8h*)(ap[mi] + 16 + akh);
#pragma unroll
            for (int i = 0; i < 8; ++i) { af[mi][i] = lo[i]; af[mi][i + 8] = hi[i]; }
        }
#pragma unroll
        for (int ni = 0; ni < 4; ++ni) {
            v8h lo = *(const v8h*)(bp[ni] + bkh);
            v8h hi = *(const v8h*)(bp[ni] + 16 + bkh);
#pragma unroll
            for (int i = 0; i < 8; ++i) { bf[ni][i] = lo[i]; bf[ni][i + 8] = hi[i]; }
        }
#pragma unroll
        for (int mi = 0; mi < 2; ++mi)
#pragma unroll
            for (int ni = 0; ni < 4; ++ni)
                acc[mi][ni] = __builtin_amdgcn_wmma_f32_16x16x32_f16(
                    false, af[mi], false, bf[ni], (short)0, acc[mi][ni], false, false);
        __syncthreads();
    }

    // epilogue: C layout — VGPR r: lanes 0-15 => M=r, lanes 16-31 => M=r+8; N = lane&15
#pragma unroll
    for (int ni = 0; ni < 4; ++ni) {
        const int nl = waveN * 64 + ni * 16 + (lane & 15);
        const int j  = n0 + nl;
        if (j >= V_OUT) continue;
        const float bj = nbias[nl];
        const int   iv = inv[j];
#pragma unroll
        for (int mi = 0; mi < 2; ++mi) {
#pragma unroll
            for (int r = 0; r < 8; ++r) {
                int row = m0 + waveM * 32 + mi * 16 + ((lane >> 4) << 3) + r;
                float s = acc[mi][ni][r] + bj;
                size_t o = (size_t)row * V_OUT + j;
                gen_masked[o] = s - 1.0e6f * (float)actionmask[o];
                float p = (iv >= 0) ? inpdist[(size_t)row * V_INP + iv] : 0.0f;
                out_scores[o] = s + p;
            }
        }
    }
}

// ---------------- softmax (online, two kernels, float4) --------------------

__global__ __launch_bounds__(256)
void k_rowstats(const float* __restrict__ S, float* __restrict__ stats) {
    __shared__ float sm[256], sl[256];
    const int b = blockIdx.x, tid = threadIdx.x;
    const float4* row4 = (const float4*)(S + (size_t)b * V_OUT);
    float m = -3.4e38f, l = 0.0f;
    for (int j = tid; j < V_OUT / 4; j += 256) {
        float4 v = row4[j];
        float m4 = fmaxf(fmaxf(v.x, v.y), fmaxf(v.z, v.w));
        float nm = fmaxf(m, m4);
        l = l * __expf(m - nm) + __expf(v.x - nm) + __expf(v.y - nm)
                               + __expf(v.z - nm) + __expf(v.w - nm);
        m = nm;
    }
    sm[tid] = m; sl[tid] = l;
    __syncthreads();
    for (int off = 128; off > 0; off >>= 1) {
        if (tid < off) {
            float m2 = sm[tid + off], l2 = sl[tid + off];
            float nm = fmaxf(sm[tid], m2);
            sl[tid] = sl[tid] * __expf(sm[tid] - nm) + l2 * __expf(m2 - nm);
            sm[tid] = nm;
        }
        __syncthreads();
    }
    if (tid == 0) { stats[2 * b] = sm[0]; stats[2 * b + 1] = sl[0]; }
}

__global__ void k_norm(float* __restrict__ S, const float* __restrict__ stats) {
    long long i = (long long)blockIdx.x * blockDim.x + threadIdx.x;   // float4 index
    const long long total4 = (long long)B_DIM * V_OUT / 4;
    if (i < total4) {
        int b = (int)((i * 4) / V_OUT);       // V_OUT % 4 == 0 -> whole float4 in one row
        float mb = stats[2 * b];
        float il = 1.0f / stats[2 * b + 1];
        float4* S4 = (float4*)S;
        float4 v = S4[i];
        v.x = __expf(v.x - mb) * il;
        v.y = __expf(v.y - mb) * il;
        v.z = __expf(v.z - mb) * il;
        v.w = __expf(v.w - mb) * il;
        S4[i] = v;
    }
}

// ---------------- launcher -------------------------------------------------

extern "C" void kernel_launch(void* const* d_in, const int* in_sizes, int n_in,
                              void* d_out, int out_size, void* d_ws, size_t ws_size,
                              hipStream_t stream) {
    const float* x     = (const float*)d_in[0];
    const float* W     = (const float*)d_in[1];
    const float* bias  = (const float*)d_in[2];
    const float* attn  = (const float*)d_in[3];
    const int*   ctx   = (const int*)d_in[4];
    const int*   i2a   = (const int*)d_in[5];
    const int*   omap  = (const int*)d_in[6];
    const int*   amask = (const int*)d_in[7];

    float* out_probs  = (float*)d_out;                               // (B, V_OUT)
    float* gen_masked = out_probs + (size_t)B_DIM * V_OUT;           // (B, V_OUT)

    // workspace: inv (V_OUT ints) | inpdist (B*V_INP f32, 125 MiB) | stats (2*B f32)
    int*   inv     = (int*)d_ws;
    size_t inv_bytes = (((size_t)V_OUT * sizeof(int)) + 255) & ~(size_t)255;
    float* inpdist = (float*)((char*)d_ws + inv_bytes);
    float* stats   = inpdist + (size_t)B_DIM * V_INP;

    k_init_inv   <<<(V_OUT + 255) / 256, 256, 0, stream>>>(inv);
    k_scatter_inv<<<(V_INP + 255) / 256, 256, 0, stream>>>(i2a, inv);
    k_zero       <<<4096, 256, 0, stream>>>(inpdist, (long long)B_DIM * V_INP);
    k_scatter_add<<<(B_DIM * SRC + 255) / 256, 256, 0, stream>>>(attn, ctx, inpdist);

    // M fastest-varying -> consecutive blocks share the same gathered W tile in L2
    dim3 grid(B_DIM / BM, (V_OUT + BN - 1) / BN);
    k_gemm<<<grid, 256, 0, stream>>>(x, W, bias, omap, amask, inv, inpdist,
                                     out_probs, gen_masked);

    k_rowstats<<<B_DIM, 256, 0, stream>>>(out_probs, stats);
    long long total4 = (long long)B_DIM * V_OUT / 4;
    k_norm<<<(int)((total4 + 255) / 256), 256, 0, stream>>>(out_probs, stats);
}
